// ProtoPN_18657337934662
// MI455X (gfx1250) — matrix operand order repaired and checked by reading, not compiled, not run on hardware
//
#include <hip/hip_runtime.h>
#include <hip/hip_bf16.h>
#include <math.h>

typedef float  vf2   __attribute__((ext_vector_type(2)));
typedef float  vf8   __attribute__((ext_vector_type(8)));
typedef __bf16 vbf4  __attribute__((ext_vector_type(4)));
typedef __bf16 vbf8  __attribute__((ext_vector_type(8)));
typedef __bf16 vbf16 __attribute__((ext_vector_type(16)));
typedef unsigned int v4u __attribute__((ext_vector_type(4)));
typedef int          v4i __attribute__((ext_vector_type(4)));
typedef int          v8i __attribute__((ext_vector_type(8)));

#define DIM   1024
#define NCLS  1000
#define NCP   1024          // classes padded to 16-multiple (zero rows, unit diag in Gram)
#define NQ    16384
#define NS    5000
#define NSP   5008          // 313 * 16
#define TOPK  5

// ---------------------------------------------------------------------------
// Row L2-normalize: one block per row, 256 threads * 4 floats = 1024 dims.
// eps_mode 0: x / max(||x||, 1e-12)   (jnp _l2n)
// eps_mode 1: x / (||x|| + 1e-6)      (ref_weight / C normalization)
// rows >= rows_valid are written as zeros (padding).
// ---------------------------------------------------------------------------
__global__ __launch_bounds__(256) void rownorm_kernel(
    const float* __restrict__ in, float* __restrict__ out32,
    __bf16* __restrict__ outbf, int rows_valid, int eps_mode)
{
  const int row = blockIdx.x;
  const int tid = threadIdx.x;
  __shared__ float red[8];
  float4 v = make_float4(0.f, 0.f, 0.f, 0.f);
  const size_t base = (size_t)row * DIM + tid * 4;
  if (row < rows_valid) v = *(const float4*)(in + base);
  float ss = v.x*v.x + v.y*v.y + v.z*v.z + v.w*v.w;
  ss += __shfl_down(ss, 16); ss += __shfl_down(ss, 8); ss += __shfl_down(ss, 4);
  ss += __shfl_down(ss, 2);  ss += __shfl_down(ss, 1);
  if ((tid & 31) == 0) red[tid >> 5] = ss;
  __syncthreads();
  if (tid == 0) { float s = 0.f; for (int i = 0; i < 8; ++i) s += red[i]; red[0] = sqrtf(s); }
  __syncthreads();
  const float nrm = red[0];
  float scale = (eps_mode == 0) ? (1.f / fmaxf(nrm, 1e-12f)) : (1.f / (nrm + 1e-6f));
  if (row >= rows_valid) scale = 0.f;
  const float4 o = make_float4(v.x*scale, v.y*scale, v.z*scale, v.w*scale);
  if (out32) *(float4*)(out32 + base) = o;
  if (outbf) {
    vbf4 b; b[0] = (__bf16)o.x; b[1] = (__bf16)o.y; b[2] = (__bf16)o.z; b[3] = (__bf16)o.w;
    *(vbf4*)(outbf + base) = b;
  }
}

// ---------------------------------------------------------------------------
// 1024x1024 transpose, 32x32 LDS tiles.
// ---------------------------------------------------------------------------
__global__ void transpose1024(const float* __restrict__ in, float* __restrict__ out)
{
  __shared__ float tile[32][33];
  const int bx = blockIdx.x * 32, by = blockIdx.y * 32;
  for (int r = threadIdx.y; r < 32; r += 8)
    tile[r][threadIdx.x] = in[(size_t)(by + r) * DIM + bx + threadIdx.x];
  __syncthreads();
  for (int r = threadIdx.y; r < 32; r += 8)
    out[(size_t)(bx + r) * DIM + by + threadIdx.x] = tile[threadIdx.x][r];
}

// ---------------------------------------------------------------------------
// FP32 WMMA NT GEMM: C[m,n] = sum_k A[m,k] * B[n,k]  (both row-major).
// One 16x16 tile per wave, v_wmma_f32_16x16x4_f32, K multiple of 8.
// A frag (16x4 f32): lane<16 -> row=lane, K={k,k+1}; lane>=16 -> K={k+2,k+3}.
// B frag mirrors A with B stored as Bt (N x K): same float2 pattern.
// Stores guarded by nvalid columns (logits has ldc=1000).
// ---------------------------------------------------------------------------
__global__ __launch_bounds__(256) void gemm_nt_f32(
    const float* __restrict__ A, int lda,
    const float* __restrict__ B, int ldb,
    float* __restrict__ C, int ldc,
    int Mt, int Nt, int K, int nvalid)
{
  const int wid  = threadIdx.x >> 5;
  const int lane = threadIdx.x & 31;
  const int tid  = blockIdx.x * 8 + wid;
  if (tid >= Mt * Nt) return;                 // whole wave exits: EXEC stays all-ones
  const int mt = tid / Nt, nt = tid % Nt;
  const int half = lane >> 4, r16 = lane & 15;
  const float* Ap = A + (size_t)(mt * 16 + r16) * lda + half * 2;
  const float* Bp = B + (size_t)(nt * 16 + r16) * ldb + half * 2;
  vf8 acc = {0.f, 0.f, 0.f, 0.f, 0.f, 0.f, 0.f, 0.f};
  for (int k = 0; k < K; k += 8) {
    vf2 a0 = *(const vf2*)(Ap + k);
    vf2 b0 = *(const vf2*)(Bp + k);
    vf2 a1 = *(const vf2*)(Ap + k + 4);
    vf2 b1 = *(const vf2*)(Bp + k + 4);
    acc = __builtin_amdgcn_wmma_f32_16x16x4_f32(false, a0, false, b0, (short)0, acc, false, false);
    acc = __builtin_amdgcn_wmma_f32_16x16x4_f32(false, a1, false, b1, (short)0, acc, false, false);
  }
  const int col = nt * 16 + r16;
  if (col < nvalid) {
    const int rbase = mt * 16 + half * 8;     // D: VGPR r -> M = r (+8 for lanes 16-31)
    #pragma unroll
    for (int r = 0; r < 8; ++r)
      C[(size_t)(rbase + r) * ldc + col] = acc[r];
  }
}

// Make padded Gram block non-singular: G[i,i] = 1 for i >= NCLS.
__global__ void fix_diag(float* __restrict__ G)
{
  const int i = NCLS + threadIdx.x;
  if (i < NCP) G[(size_t)i * NCP + i] = 1.f;
}

// ---------------------------------------------------------------------------
// In-place lower Cholesky of G (NCP x NCP, SPD, kappa ~ 3e4 -> fp32 OK).
// Single 1024-thread workgroup; upper triangle left untouched.
// ---------------------------------------------------------------------------
__global__ __launch_bounds__(1024) void cholesky_kernel(float* __restrict__ G)
{
  const int tid = threadIdx.x;
  for (int k = 0; k < NCP; ++k) {
    if (tid == 0) G[(size_t)k * NCP + k] = sqrtf(G[(size_t)k * NCP + k]);
    __syncthreads();
    const float inv = 1.0f / G[(size_t)k * NCP + k];
    for (int i = k + 1 + tid; i < NCP; i += 1024)
      G[(size_t)i * NCP + k] *= inv;
    __syncthreads();
    for (int i = k + 1 + tid; i < NCP; i += 1024) {
      const float lik = G[(size_t)i * NCP + k];
      float* gi = G + (size_t)i * NCP;
      for (int j = k + 1; j <= i; ++j)
        gi[j] -= lik * G[(size_t)j * NCP + k];
    }
    __syncthreads();
  }
}

// Forward solve L * Y = Rn. Thread j owns column j -> no barriers needed.
// Y (and later X) are L2-resident (4 MB), so the row re-reads stay in L2.
__global__ __launch_bounds__(128) void trsv_fwd(
    const float* __restrict__ G, const float* __restrict__ Rn, float* __restrict__ XY)
{
  const int j = blockIdx.x * blockDim.x + threadIdx.x;
  for (int i = 0; i < NCP; ++i) {
    float s = Rn[(size_t)i * NCP + j];
    const float* Li = G + (size_t)i * NCP;
    for (int p = 0; p < i; ++p) s -= Li[p] * XY[(size_t)p * NCP + j];
    XY[(size_t)i * NCP + j] = s / Li[i];
  }
}

// Backward solve L^T * X = Y, in place (column-private).
__global__ __launch_bounds__(128) void trsv_bwd(
    const float* __restrict__ G, float* __restrict__ XY)
{
  const int j = blockIdx.x * blockDim.x + threadIdx.x;
  for (int i = NCP - 1; i >= 0; --i) {
    float s = XY[(size_t)i * NCP + j];
    for (int p = i + 1; p < NCP; ++p)
      s -= G[(size_t)p * NCP + i] * XY[(size_t)p * NCP + j];
    XY[(size_t)i * NCP + j] = s / G[(size_t)i * NCP + i];
  }
}

// ---------------------------------------------------------------------------
// sim = s_n @ q_n^T (bf16 WMMA) fused with per-row top-5 over the q axis.
// Block: 16 s_n rows staged into LDS (32 KB, one contiguous block) via the
// Tensor Data Mover when available (1-D D#: data_size=8B, tile_dim0=4096),
// issued by wave 0 and synchronized with s_wait_tensorcnt + barrier.
// 8 waves stripe the 1024 q-tiles; qtn_bf (32 MB) is L2-resident so B comes
// straight from global. Each 16x16 f32 tile is spilled through a per-wave LDS
// buffer so lane L can scan row L%16 / column-half L/16; per-lane running
// top-5 in registers; final 80-candidate merge per row via LDS.
// ---------------------------------------------------------------------------
__global__ __launch_bounds__(256) void sim_topk_kernel(
    const __bf16* __restrict__ S, const __bf16* __restrict__ Q,
    int* __restrict__ topk)
{
  __shared__ __align__(16) __bf16 Atile[16 * DIM];   // 32 KB
  __shared__ float tilebuf[8][16 * 17];
  __shared__ float mv[16][80];
  __shared__ int   mi[16][80];

  const int m0 = blockIdx.x * 16;
#if __has_builtin(__builtin_amdgcn_tensor_load_to_lds)
  if (threadIdx.x < 32) {   // wave 0 issues one TDM transfer for the whole tile
    const unsigned long long ga = (unsigned long long)(const void*)(S + (size_t)m0 * DIM);
    // ISA 10.2: inside the LDS aperture, flat addr[31:0] IS the LDS address.
    const unsigned lds_addr = (unsigned)(unsigned long long)(const void*)Atile;
    const unsigned td0 = 4096u;  // 32768 B / 8 B elements (data_size code 3)
    v4u g0 = { 1u,                                   // count=1, user mode
               lds_addr,
               (unsigned)(ga & 0xffffffffull),
               (unsigned)((ga >> 32) & 0x1ffffffull) | (2u << 30) };  // type=2
    v8i g1 = { (int)(3u << 16),                      // data_size = 8B; mask=0
               (int)((td0 & 0xffffu) << 16),         // tensor_dim0[15:0]
               (int)((td0 >> 16) | (1u << 16)),      // tensor_dim0[31:16], tensor_dim1=1
               (int)((td0 & 0xffffu) << 16),         // tensor_dim1[31:16]=0, tile_dim0=4096
               0,                                    // tile_dim1=0 (unused), tile_dim2=0
               (int)td0,                             // tensor_dim0_stride[31:0]
               0, 0 };                               // stride hi, dim1_stride
    v4i gz = { 0, 0, 0, 0 };
#if __clang_major__ >= 23
    v8i gz8 = { 0, 0, 0, 0, 0, 0, 0, 0 };
    __builtin_amdgcn_tensor_load_to_lds(g0, g1, gz, gz, gz8, 0);
#else
    __builtin_amdgcn_tensor_load_to_lds(g0, g1, gz, gz, 0);
#endif
    __builtin_amdgcn_s_wait_tensorcnt(0);
  }
#else
  for (int i = threadIdx.x; i < 16 * DIM / 8; i += 256)
    ((float4*)Atile)[i] = ((const float4*)(S + (size_t)m0 * DIM))[i];
#endif
  __syncthreads();

  const int wid = threadIdx.x >> 5, lane = threadIdx.x & 31;
  const int r16 = lane & 15, half = lane >> 4;

  float tv[TOPK]; int ti[TOPK];
  #pragma unroll
  for (int j = 0; j < TOPK; ++j) { tv[j] = -3e38f; ti[j] = 0x7fffffff; }

  // bf16 A frag: lane<16 rows, K {k..k+7} then {k+16..k+23}; lanes>=16 offset +8.
  const __bf16* Ap = Atile + (size_t)r16 * DIM + half * 8;

  for (int qt = wid; qt < NQ / 16; qt += 8) {
    const int n0 = qt * 16;
    // bf16 B frag from Bt rows: lane<16 K {k..k+7} then {k+16..k+23}; lanes>=16 offset +8.
    const __bf16* Bp = Q + (size_t)(n0 + r16) * DIM + half * 8;
    if (qt + 8 < NQ / 16)
      __builtin_prefetch(Q + (size_t)((qt + 8) * 16 + r16) * DIM + half * 16, 0, 1);
    vf8 acc = {0.f, 0.f, 0.f, 0.f, 0.f, 0.f, 0.f, 0.f};
    for (int k = 0; k < DIM; k += 32) {
      vbf16 a, b;
      *(vbf8*)&a         = *(const vbf8*)(Ap + k);
      *((vbf8*)&a + 1)   = *(const vbf8*)(Ap + k + 16);
      *(vbf8*)&b         = *(const vbf8*)(Bp + k);
      *((vbf8*)&b + 1)   = *(const vbf8*)(Bp + k + 16);
      acc = __builtin_amdgcn_wmma_f32_16x16x32_bf16(false, a, false, b, (short)0, acc,
                                                    false, false);
    }
    #pragma unroll
    for (int r = 0; r < 8; ++r)                       // D: (M = r + half*8, N = r16)
      tilebuf[wid][(half * 8 + r) * 17 + r16] = acc[r];
    asm volatile("s_wait_dscnt 0" ::: "memory");      // wave-local store->load ordering
    #pragma unroll
    for (int c = 0; c < 8; ++c) {
      const float v  = tilebuf[wid][r16 * 17 + half * 8 + c];
      const int   qi = n0 + half * 8 + c;
      if (v > tv[TOPK - 1] || (v == tv[TOPK - 1] && qi < ti[TOPK - 1])) {
        tv[TOPK - 1] = v; ti[TOPK - 1] = qi;
        #pragma unroll
        for (int s = TOPK - 1; s > 0; --s) {
          if (tv[s] > tv[s - 1] || (tv[s] == tv[s - 1] && ti[s] < ti[s - 1])) {
            const float fv = tv[s]; tv[s] = tv[s - 1]; tv[s - 1] = fv;
            const int   fi = ti[s]; ti[s] = ti[s - 1]; ti[s - 1] = fi;
          }
        }
      }
    }
  }

  const int slot = (wid * 2 + half) * TOPK;
  #pragma unroll
  for (int j = 0; j < TOPK; ++j) { mv[r16][slot + j] = tv[j]; mi[r16][slot + j] = ti[j]; }
  __syncthreads();
  if (threadIdx.x < 16) {
    const int row = threadIdx.x, grow = m0 + row;
    if (grow < NS) {
      for (int j = 0; j < TOPK; ++j) {
        float bv = -3e38f; int bi = 0x7fffffff, bs = 0;
        for (int s = 0; s < 80; ++s) {
          const float v = mv[row][s]; const int id = mi[row][s];
          if (v > bv || (v == bv && id < bi)) { bv = v; bi = id; bs = s; }
        }
        mv[row][bs] = -3.3e38f;
        topk[grow * TOPK + j] = bi;
      }
    }
  }
}

// ---------------------------------------------------------------------------
// protos_mean: mean of [protos_t[c], samples_t[c,0..4], q_t[topk[c,0..24]]]
// (31 rows) then L2-normalize. One block per class; pad classes write zeros.
// ---------------------------------------------------------------------------
__global__ __launch_bounds__(256) void protomean_kernel(
    const float* __restrict__ pt, const float* __restrict__ st,
    const float* __restrict__ qt, const int* __restrict__ topk,
    float* __restrict__ pm)
{
  const int c = blockIdx.x, tid = threadIdx.x;
  __shared__ float red[8];
  float ax = 0.f, ay = 0.f, az = 0.f, aw = 0.f;
  const int d = tid * 4;
  if (c < NCLS) {
    float4 p = *(const float4*)(pt + (size_t)c * DIM + d);
    ax = p.x; ay = p.y; az = p.z; aw = p.w;
    for (int k = 0; k < 5; ++k) {
      const float4 s = *(const float4*)(st + (size_t)(c * 5 + k) * DIM + d);
      ax += s.x; ay += s.y; az += s.z; aw += s.w;
    }
    for (int t = 0; t < 25; ++t) {
      const int q = topk[c * 25 + t];          // (c*5+k)*5 + j, scalar per block
      const float4 s = *(const float4*)(qt + (size_t)q * DIM + d);
      ax += s.x; ay += s.y; az += s.z; aw += s.w;
    }
    const float inv31 = 1.f / 31.f;
    ax *= inv31; ay *= inv31; az *= inv31; aw *= inv31;
  }
  float ss = ax * ax + ay * ay + az * az + aw * aw;
  ss += __shfl_down(ss, 16); ss += __shfl_down(ss, 8); ss += __shfl_down(ss, 4);
  ss += __shfl_down(ss, 2);  ss += __shfl_down(ss, 1);
  if ((tid & 31) == 0) red[tid >> 5] = ss;
  __syncthreads();
  if (tid == 0) { float s = 0.f; for (int i = 0; i < 8; ++i) s += red[i]; red[0] = sqrtf(s); }
  __syncthreads();
  float scale = 1.f / fmaxf(red[0], 1e-12f);
  if (c >= NCLS) scale = 0.f;
  *(float4*)(pm + (size_t)c * DIM + d) =
      make_float4(ax * scale, ay * scale, az * scale, aw * scale);
}

// ---------------------------------------------------------------------------
// Launch pipeline. Scalars session=9, way=100 (n_cls=1000) per setup_inputs.
// ---------------------------------------------------------------------------
extern "C" void kernel_launch(void* const* d_in, const int* in_sizes, int n_in,
                              void* d_out, int out_size, void* d_ws, size_t ws_size,
                              hipStream_t stream)
{
  (void)in_sizes; (void)n_in; (void)out_size; (void)ws_size;
  const float* audio = (const float*)d_in[0];   // 16384 x 1024
  const float* oldp  = (const float*)d_in[1];   // 1000 x 1024
  const float* mem   = (const float*)d_in[2];   // 1000 x 5 x 1024
  const float* refw  = (const float*)d_in[3];   // 1000 x 1024

  char* w = (char*)d_ws;
  size_t off = 0;
  auto take = [&](size_t bytes) -> char* {
    char* p = w + off; off = (off + bytes + 255) & ~(size_t)255; return p;
  };
  float*  qn     = (float*)take((size_t)NQ  * DIM * 4);   // q normalized; reused as q_t normalized
  float*  qt     = (float*)take((size_t)NQ  * DIM * 4);   // q @ M
  __bf16* qtn_bf = (__bf16*)take((size_t)NQ * DIM * 2);
  float*  sn     = (float*)take((size_t)NSP * DIM * 4);   // normalized memory samples (padded)
  float*  st     = (float*)take((size_t)NSP * DIM * 4);   // samples @ M
  __bf16* stn_bf = (__bf16*)take((size_t)NSP * DIM * 2);
  float*  Pn     = (float*)take((size_t)NCP * DIM * 4);   // l2n(old_protos)
  float*  Cn     = (float*)take((size_t)NCP * DIM * 4);   // Pn / (||Pn|| + 1e-6)
  float*  Rn     = (float*)take((size_t)NCP * DIM * 4);
  float*  G      = (float*)take((size_t)NCP * NCP * 4);   // Gram, then L in lower triangle
  float*  XY     = (float*)take((size_t)NCP * DIM * 4);   // Y then X = G^-1 Rn
  float*  Xt     = (float*)take((size_t)NCP * DIM * 4);
  float*  CnT    = (float*)take((size_t)NCP * DIM * 4);
  float*  Mtb    = (float*)take((size_t)DIM * DIM * 4);   // M^T
  float*  pt     = (float*)take((size_t)NCP * DIM * 4);   // protos_t
  float*  pm     = (float*)take((size_t)NCP * DIM * 4);   // protos_mean normalized
  int*    topk   = (int*)take((size_t)NSP * TOPK * 4);

  auto gg = [](int tiles) { return (tiles + 7) / 8; };
  const dim3 tg(32, 32), tb(32, 8);

  // 1) normalizations
  rownorm_kernel<<<NQ,  256, 0, stream>>>(audio, qn, nullptr, NQ,   0);
  rownorm_kernel<<<NCP, 256, 0, stream>>>(oldp,  Pn, nullptr, NCLS, 0);
  rownorm_kernel<<<NCP, 256, 0, stream>>>(Pn,    Cn, nullptr, NCLS, 1);
  rownorm_kernel<<<NCP, 256, 0, stream>>>(refw,  Rn, nullptr, NCLS, 1);
  rownorm_kernel<<<NSP, 256, 0, stream>>>(mem,   sn, nullptr, NS,   0);

  // 2) M^T = X^T Cn^T with X = (Cn Cn^T)^-1 Rn   (all fp32 WMMA / fp32 solves)
  gemm_nt_f32<<<gg(64 * 64), 256, 0, stream>>>(Cn, DIM, Cn, DIM, G, NCP, 64, 64, DIM, NCP);
  fix_diag<<<1, 32, 0, stream>>>(G);
  cholesky_kernel<<<1, 1024, 0, stream>>>(G);
  trsv_fwd<<<8, 128, 0, stream>>>(G, Rn, XY);
  trsv_bwd<<<8, 128, 0, stream>>>(G, XY);
  transpose1024<<<tg, tb, 0, stream>>>(XY, Xt);
  transpose1024<<<tg, tb, 0, stream>>>(Cn, CnT);
  gemm_nt_f32<<<gg(64 * 64), 256, 0, stream>>>(Xt, DIM, CnT, DIM, Mtb, DIM, 64, 64, NCP, DIM);

  // 3) projections through M (fp32 WMMA)
  gemm_nt_f32<<<gg((NQ / 16) * 64), 256, 0, stream>>>(qn, DIM, Mtb, DIM, qt, DIM, NQ / 16, 64, DIM, DIM);
  gemm_nt_f32<<<gg((NSP / 16) * 64), 256, 0, stream>>>(sn, DIM, Mtb, DIM, st, DIM, NSP / 16, 64, DIM, DIM);
  gemm_nt_f32<<<gg(64 * 64), 256, 0, stream>>>(Pn, DIM, Mtb, DIM, pt, DIM, 64, 64, DIM, DIM);

  // 4) re-normalize projected vectors; bf16 copies for the ranking GEMM
  rownorm_kernel<<<NQ,  256, 0, stream>>>(qt, qn, qtn_bf, NQ, 0);     // qn now = l2n(q_t)
  rownorm_kernel<<<NSP, 256, 0, stream>>>(st, nullptr, stn_bf, NS, 0);

  // 5) 168-GFLOP sim GEMM (bf16 WMMA, TDM-staged A tile) fused with top-5
  sim_topk_kernel<<<NSP / 16, 256, 0, stream>>>(stn_bf, qtn_bf, topk);

  // 6) gather + mean + normalize prototypes
  protomean_kernel<<<NCP, 256, 0, stream>>>(pt, st, qt, topk, pm);

  // 7) logits = l2n(q_t) @ protos_mean^T  -> d_out (16384 x 1000, ldc = 1000)
  gemm_nt_f32<<<gg((NQ / 16) * 64), 256, 0, stream>>>(qn, DIM, pm, DIM,
                                                      (float*)d_out, NCLS,
                                                      NQ / 16, 64, DIM, NCLS);
}
